// AnchorAttention_893353198012
// MI455X (gfx1250) — matrix operand order repaired and checked by reading, not compiled, not run on hardware
//
#include <hip/hip_runtime.h>
#include <cstdint>

// ---------------------------------------------------------------- constants
constexpr int Bc   = 8;
constexpr int Nn   = 4096;
constexpr int DIMc = 1024;
constexpr int H    = 16;
constexpr int DH   = 64;
constexpr int ANC  = 64;
constexpr int TI   = 2048;           // 2*INNER columns of W_emb
constexpr float SCALEF = 0.125f;     // 64^-0.5

typedef __attribute__((ext_vector_type(16))) __bf16 v16bf;
typedef __attribute__((ext_vector_type(8)))  float  v8f;
typedef __attribute__((ext_vector_type(4)))  unsigned int u32x4;
typedef __attribute__((ext_vector_type(8)))  unsigned int u32x8;

#define DEV static __device__ __forceinline__

// ---------------------------------------------------------------- WMMA core
DEV v8f wmma_bf16(v16bf a, v16bf b, v8f c) {
  return __builtin_amdgcn_wmma_f32_16x16x32_bf16(false, a, false, b,
                                                 (short)0, c, false, false);
}

// ---------------------------------------------------------------- async paths
DEV uint32_t lds_off(const void* p) { return (uint32_t)(uintptr_t)p; }

// One 16B chunk, memory -> LDS, no VGPR data round trip (ASYNCcnt).
DEV void async_b128(void* lds, const void* gaddr) {
  asm volatile("global_load_async_to_lds_b128 %0, %1, off"
               :: "v"(lds_off(lds)), "v"((unsigned long long)(uintptr_t)gaddr)
               : "memory");
}
DEV void wait_async0() { asm volatile("s_wait_asynccnt 0x0" ::: "memory"); }

// TDM: 2-D bf16 tile (tile0 x tile1, row stride stride0 elements) -> LDS.
// D# group0/group1 packed per CDNA5 ISA 8.3/8.4; groups 2/3 omitted (2-D).
DEV void tdm_load_2d(const void* gptr, void* ldsptr,
                     uint32_t dim0, uint32_t dim1,
                     uint32_t tile0, uint32_t tile1, uint32_t stride0) {
  uint64_t ga = (uint64_t)(uintptr_t)gptr;
  u32x4 g0;
  g0[0] = 1u;                                           // count=1, no gather
  g0[1] = lds_off(ldsptr);                              // lds_addr
  g0[2] = (uint32_t)ga;                                 // global_addr[31:0]
  g0[3] = (uint32_t)((ga >> 32) & 0x01FFFFFFu) | 0x80000000u; // addr[56:32]|type=2
  u32x8 g1;
  g1[0] = 1u << 16;                                     // data_size = 2 bytes
  g1[1] = (dim0 & 0xFFFFu) << 16;                       // tensor_dim0[15:0]
  g1[2] = (dim0 >> 16) | ((dim1 & 0xFFFFu) << 16);      // dim0 hi | dim1 lo
  g1[3] = (dim1 >> 16) | (tile0 << 16);                 // dim1 hi | tile_dim0
  g1[4] = tile1;                                        // tile_dim1, tile_dim2=0
  g1[5] = stride0;                                      // tensor_dim0_stride lo
  g1[6] = 0u;
  g1[7] = 0u;
  asm volatile("tensor_load_to_lds %0, %1"
               :: "s"(g0), "s"(g1)
               : "memory");
}
DEV void wait_tensor0() { __builtin_amdgcn_s_wait_tensorcnt(0); }

// ---------------------------------------------------------------- fragments
// A 16x32, source row-major [m][k].  (contiguous k = 16*half + e)
DEV v16bf frag_a_row(const __bf16* base, int stride, int lane) {
  int m = lane & 15, half = lane >> 4;
  v16bf f;
#pragma unroll
  for (int e = 0; e < 16; ++e) {
    int k = (half << 4) + e;
    f[e] = base[m * stride + k];
  }
  return f;
}

// A 16x32 from a [k][m] source: implicit transpose.
DEV v16bf frag_a_col(const __bf16* base, int stride, int lane) {
  int m = lane & 15, half = lane >> 4;
  v16bf f;
#pragma unroll
  for (int e = 0; e < 16; ++e) {
    int k = (half << 4) + e;
    f[e] = base[k * stride + m];
  }
  return f;
}

// B 32x16, source stored [k][n] (k-major).
DEV v16bf frag_b_kmajor(const __bf16* base, int stride, int lane) {
  int n = lane & 15, kb = (lane >> 4) << 4;
  v16bf f;
#pragma unroll
  for (int e = 0; e < 16; ++e) f[e] = base[(kb + e) * stride + n];
  return f;
}

// B 32x16, source stored [n][k] (n-major) -> contiguous 16B runs.
DEV v16bf frag_b_nmajor(const __bf16* base, int stride, int lane) {
  int n = lane & 15, kb = (lane >> 4) << 4;
  v16bf f;
#pragma unroll
  for (int e = 0; e < 16; ++e) f[e] = base[n * stride + kb + e];
  return f;
}

// ---------------------------------------------------------------- weight cvt
__global__ __launch_bounds__(256) void k_cvt(const float* __restrict__ s,
                                             __bf16* __restrict__ d, int n) {
  for (int i = blockIdx.x * 256 + threadIdx.x; i < n; i += gridDim.x * 256)
    d[i] = (__bf16)s[i];
}

// ---------------------------------------------------------------- stage 1
// emb = x @ W_emb^T, fused anchor projection + softmax on the qk half.
// grid (256,32); block 256 = 8 waves.  BM=128, BN=64 (one head), BK=32.
__global__ __launch_bounds__(256)
void k_embed(const float* __restrict__ x, const __bf16* __restrict__ Wemb,
             const __bf16* __restrict__ Wanc, __bf16* __restrict__ attnG,
             __bf16* __restrict__ vG, float* __restrict__ sdeg) {
  __shared__ __align__(16) __bf16 As[2][128 * 32];  // [m][k]
  __shared__ __align__(16) __bf16 Bs[2][64 * 32];   // [n][k]
  __shared__ __align__(16) __bf16 Qs[128 * 64];     // qk tile for anchor stage

  const int tid  = threadIdx.x;
  const int lane = tid & 31;
  const int wave = tid >> 5;
  const int r0w  = wave * 16;
  const int row0 = blockIdx.x * 128;                // token-row base (b*N + n)
  const int col0 = blockIdx.y * 64;                 // emb column base
  const size_t rowBase = (size_t)row0 * DIMc;

  auto load_tiles = [&](int k0, int bsel) {
#pragma unroll
    for (int j = 0; j < 4; ++j) {                   // A: fp32 -> bf16 via VGPRs
      int q = tid + j * 256;
      int r = q >> 3, kq = (q & 7) << 2;
      const float4 v4 = *(const float4*)(x + rowBase + (size_t)r * DIMc + k0 + kq);
      __bf16* p = &As[bsel][r * 32 + kq];
      p[0] = (__bf16)v4.x; p[1] = (__bf16)v4.y;
      p[2] = (__bf16)v4.z; p[3] = (__bf16)v4.w;
    }
    // B: 256 x 16B chunks, async direct-to-LDS
    int n = tid >> 2, kc = (tid & 3) << 3;
    async_b128(&Bs[bsel][n * 32 + kc], &Wemb[(size_t)(col0 + n) * DIMc + k0 + kc]);
  };

  v8f zero = {0.f, 0.f, 0.f, 0.f, 0.f, 0.f, 0.f, 0.f};
  v8f acc[4] = {zero, zero, zero, zero};

  load_tiles(0, 0);
  wait_async0();
  __syncthreads();
  int buf = 0;
  for (int kt = 0; kt < DIMc / 32; ++kt) {
    if (kt + 1 < DIMc / 32) load_tiles((kt + 1) * 32, buf ^ 1);
    const __bf16* Ab = As[buf];
    const __bf16* Bb = Bs[buf];
    v16bf af = frag_a_row(Ab + r0w * 32, 32, lane);
#pragma unroll
    for (int t = 0; t < 4; ++t) {
      v16bf bf_ = frag_b_nmajor(Bb + (t * 16) * 32, 32, lane);
      acc[t] = wmma_bf16(af, bf_, acc[t]);
    }
    wait_async0();
    __syncthreads();
    buf ^= 1;
  }

  const int half = lane >> 4, nc = lane & 15;
  const int bidx = row0 / Nn;
  const int nb   = row0 % Nn;

  if (blockIdx.y >= 16) {
    // --------- v half: store bf16 into [b,h,n,d]
    const int hv = blockIdx.y - 16;
    const size_t vbase = ((size_t)(bidx * H + hv) * Nn + nb) * DH;
#pragma unroll
    for (int t = 0; t < 4; ++t)
#pragma unroll
      for (int r = 0; r < 8; ++r) {
        int row = r0w + r + 8 * half;
        vG[vbase + (size_t)row * DH + t * 16 + nc] = (__bf16)acc[t][r];
      }
    return;
  }

  // --------- qk half: stage tile, anchor projection (2nd WMMA stage), softmax
  const int h = blockIdx.y;
#pragma unroll
  for (int t = 0; t < 4; ++t)
#pragma unroll
    for (int r = 0; r < 8; ++r) {
      int row = r0w + r + 8 * half;
      Qs[row * 64 + t * 16 + nc] = (__bf16)acc[t][r];
    }
  __syncthreads();

  v8f lacc[4] = {zero, zero, zero, zero};
#pragma unroll
  for (int kk = 0; kk < 64; kk += 32) {
    v16bf af = frag_a_row(Qs + r0w * 64 + kk, 64, lane);
#pragma unroll
    for (int t = 0; t < 4; ++t) {
      // B[k=d][n=a] = Wanc[a][d]  (n-major source, row length 64)
      v16bf bf_ = frag_b_nmajor(Wanc + (t * 16) * 64 + kk, 64, lane);
      lacc[t] = wmma_bf16(af, bf_, lacc[t]);
    }
  }

  float pv[4][8];
#pragma unroll
  for (int t = 0; t < 4; ++t)
#pragma unroll
    for (int r = 0; r < 8; ++r) pv[t][r] = lacc[t][r] * SCALEF;

#pragma unroll
  for (int r = 0; r < 8; ++r) {
    float mx = fmaxf(fmaxf(pv[0][r], pv[1][r]), fmaxf(pv[2][r], pv[3][r]));
#pragma unroll
    for (int off = 8; off >= 1; off >>= 1) mx = fmaxf(mx, __shfl_xor(mx, off, 16));
    float sm = 0.f;
#pragma unroll
    for (int t = 0; t < 4; ++t) { pv[t][r] = __expf(pv[t][r] - mx); sm += pv[t][r]; }
#pragma unroll
    for (int off = 8; off >= 1; off >>= 1) sm += __shfl_xor(sm, off, 16);
    float inv = 1.f / sm;
#pragma unroll
    for (int t = 0; t < 4; ++t) pv[t][r] *= inv;
  }

  const size_t abase = ((size_t)(bidx * H + h) * Nn + nb) * ANC;
#pragma unroll
  for (int t = 0; t < 4; ++t) {
    int a = t * 16 + nc;
    float colsum = 0.f;
#pragma unroll
    for (int r = 0; r < 8; ++r) {
      int row = r0w + r + 8 * half;
      attnG[abase + (size_t)row * ANC + a] = (__bf16)pv[t][r];
      colsum += pv[t][r];
    }
    atomicAdd(&sdeg[(bidx * H + h) * ANC + a], colsum);
  }
}

// ---------------------------------------------------------------- stage 2
// out2[b,h,a,d] = recip[a] * sum_n attn[n,a] * v[n,d]   (M=N=64, K=4096)
// grid 128 (b*h); block 128 = 4 waves.  Tiles staged with async-to-LDS.
__global__ __launch_bounds__(128)
void k_aggregate(const __bf16* __restrict__ attnG, const __bf16* __restrict__ vG,
                 const float* __restrict__ sdeg, __bf16* __restrict__ out2) {
  __shared__ __align__(16) __bf16 AS[2][32 * 64];  // attn tile [k=n][a]
  __shared__ __align__(16) __bf16 VS[2][32 * 64];  // v tile    [k=n][d]

  const int bh   = blockIdx.x;
  const int tid  = threadIdx.x;
  const int lane = tid & 31;
  const int wave = tid >> 5;
  const int a0   = wave * 16;
  const __bf16* aP = attnG + (size_t)bh * Nn * ANC;
  const __bf16* vP = vG    + (size_t)bh * Nn * DH;

  auto load_tiles = [&](int n0, int bsel) {
#pragma unroll
    for (int j = 0; j < 2; ++j) {                   // 256 x 16B chunks per tile
      int e = tid + j * 128;
      int r = e >> 3, kc = (e & 7) << 3;
      async_b128(&AS[bsel][r * 64 + kc], &aP[(size_t)(n0 + r) * ANC + kc]);
      async_b128(&VS[bsel][r * 64 + kc], &vP[(size_t)(n0 + r) * DH + kc]);
    }
  };

  v8f zero = {0.f, 0.f, 0.f, 0.f, 0.f, 0.f, 0.f, 0.f};
  v8f acc[4] = {zero, zero, zero, zero};

  load_tiles(0, 0);
  wait_async0();
  __syncthreads();
  int buf = 0;
  for (int kt = 0; kt < Nn / 32; ++kt) {
    if (kt + 1 < Nn / 32) load_tiles((kt + 1) * 32, buf ^ 1);
    v16bf af = frag_a_col(&AS[buf][a0], 64, lane);  // A = attn^T
#pragma unroll
    for (int t = 0; t < 4; ++t) {
      v16bf bf_ = frag_b_kmajor(&VS[buf][t * 16], 64, lane);
      acc[t] = wmma_bf16(af, bf_, acc[t]);
    }
    wait_async0();
    __syncthreads();
    buf ^= 1;
  }

  const int half = lane >> 4, nc = lane & 15;
#pragma unroll
  for (int r = 0; r < 8; ++r) {
    int a = a0 + r + 8 * half;
    float sv = sdeg[bh * ANC + a];
    float rc = (sv != 0.f) ? (1.f / sv) : 0.f;      // inf -> 0 semantics
#pragma unroll
    for (int t = 0; t < 4; ++t)
      out2[((size_t)bh * ANC + a) * DH + t * 16 + nc] = (__bf16)(acc[t][r] * rc);
  }
}

// ---------------------------------------------------------------- stage 3
// y[b,n,h*64+d] = sum_a attn[b,h,n,a] * out2[b,h,a,d]   (K=64)
// Block tiles staged by the Tensor Data Mover (one wave issues 2 TDM ops).
__global__ __launch_bounds__(256)
void k_broadcast(const __bf16* __restrict__ attnG, const __bf16* __restrict__ out2,
                 __bf16* __restrict__ yG) {
  __shared__ __align__(16) __bf16 AS[128 * 64];    // attn tile [n][a]
  __shared__ __align__(16) __bf16 BS[64 * 64];     // out2 [a][d] (k-major)

  const int bh = blockIdx.x >> 5;
  const int n0 = (blockIdx.x & 31) * 128;
  const int tid = threadIdx.x, lane = tid & 31, wave = tid >> 5, r0w = wave * 16;
  const __bf16* aP = attnG + ((size_t)bh * Nn + n0) * ANC;
  const __bf16* oP = out2 + (size_t)bh * ANC * DH;

  if (wave == 0) {
    tdm_load_2d(aP, AS, /*dim0=*/64, /*dim1=*/128, /*tile0=*/64, /*tile1=*/128, /*stride0=*/64);
    tdm_load_2d(oP, BS, 64, 64, 64, 64, 64);
    wait_tensor0();
  }
  __syncthreads();

  v8f zero = {0.f, 0.f, 0.f, 0.f, 0.f, 0.f, 0.f, 0.f};
  v8f acc[4] = {zero, zero, zero, zero};
#pragma unroll
  for (int kk = 0; kk < 64; kk += 32) {
    v16bf af = frag_a_row(AS + r0w * 64 + kk, 64, lane);
#pragma unroll
    for (int t = 0; t < 4; ++t) {
      v16bf bf_ = frag_b_kmajor(&BS[kk * 64 + t * 16], 64, lane);
      acc[t] = wmma_bf16(af, bf_, acc[t]);
    }
  }

  const int half = lane >> 4, nc = lane & 15;
  const int b = bh >> 4, h = bh & 15;
#pragma unroll
  for (int t = 0; t < 4; ++t)
#pragma unroll
    for (int r = 0; r < 8; ++r) {
      int row = r0w + r + 8 * half;
      yG[((size_t)(b * Nn + n0 + row)) * 1024 + h * 64 + t * 16 + nc] = (__bf16)acc[t][r];
    }
}

// ---------------------------------------------------------------- stage 4
// out = y @ W_out^T + b_out  (fp32 output).  All staging async-to-LDS.
__global__ __launch_bounds__(256)
void k_output(const __bf16* __restrict__ yG, const __bf16* __restrict__ Wout,
              const float* __restrict__ bout, float* __restrict__ outG) {
  __shared__ __align__(16) __bf16 As[2][128 * 32]; // [m][k]
  __shared__ __align__(16) __bf16 Bs[2][64 * 32];  // [n][k]

  const int tid = threadIdx.x, lane = tid & 31, wave = tid >> 5, r0w = wave * 16;
  const int row0 = blockIdx.x * 128;
  const int col0 = blockIdx.y * 64;
  const size_t rowBase = (size_t)row0 * DIMc;

  auto load_tiles = [&](int k0, int bsel) {
#pragma unroll
    for (int j = 0; j < 2; ++j) {                  // A: 512 x 16B chunks
      int e = tid + j * 256;
      int r = e >> 2, kc = (e & 3) << 3;
      async_b128(&As[bsel][r * 32 + kc], &yG[rowBase + (size_t)r * DIMc + k0 + kc]);
    }
    int n = tid >> 2, kc = (tid & 3) << 3;         // B: 256 x 16B chunks
    async_b128(&Bs[bsel][n * 32 + kc], &Wout[(size_t)(col0 + n) * DIMc + k0 + kc]);
  };

  v8f zero = {0.f, 0.f, 0.f, 0.f, 0.f, 0.f, 0.f, 0.f};
  v8f acc[4] = {zero, zero, zero, zero};

  load_tiles(0, 0);
  wait_async0();
  __syncthreads();
  int buf = 0;
  for (int kt = 0; kt < DIMc / 32; ++kt) {
    if (kt + 1 < DIMc / 32) load_tiles((kt + 1) * 32, buf ^ 1);
    v16bf af = frag_a_row(As[buf] + r0w * 32, 32, lane);
#pragma unroll
    for (int t = 0; t < 4; ++t) {
      v16bf bf_ = frag_b_nmajor(Bs[buf] + (t * 16) * 32, 32, lane);
      acc[t] = wmma_bf16(af, bf_, acc[t]);
    }
    wait_async0();
    __syncthreads();
    buf ^= 1;
  }

  const int half = lane >> 4, nc = lane & 15;
#pragma unroll
  for (int t = 0; t < 4; ++t) {
    int col = col0 + t * 16 + nc;
    float bb = bout[col];
#pragma unroll
    for (int r = 0; r < 8; ++r) {
      int row = r0w + r + 8 * half;
      outG[(size_t)(row0 + row) * DIMc + col] = acc[t][r] + bb;
    }
  }
}

// ---------------------------------------------------------------- launcher
extern "C" void kernel_launch(void* const* d_in, const int* in_sizes, int n_in,
                              void* d_out, int out_size, void* d_ws, size_t ws_size,
                              hipStream_t stream) {
  const float* x    = (const float*)d_in[0];
  const float* Wemb = (const float*)d_in[1];
  const float* Wanc = (const float*)d_in[2];
  const float* Wout = (const float*)d_in[3];
  const float* bout = (const float*)d_in[4];
  float* out = (float*)d_out;

  char* ws = (char*)d_ws;
  size_t off = 0;
  auto take = [&](size_t bytes) {
    char* p = ws + off;
    off += (bytes + 255) & ~(size_t)255;
    return p;
  };
  __bf16* WembB = (__bf16*)take((size_t)TI * DIMc * 2);
  __bf16* WoutB = (__bf16*)take((size_t)DIMc * DIMc * 2);
  __bf16* WancB = (__bf16*)take((size_t)ANC * DH * 2);
  __bf16* attnB = (__bf16*)take((size_t)Bc * H * Nn * ANC * 2);
  __bf16* vB    = (__bf16*)take((size_t)Bc * H * Nn * DH * 2);
  float*  sdeg  = (float*)take((size_t)Bc * H * ANC * 4);
  __bf16* out2B = (__bf16*)take((size_t)Bc * H * ANC * DH * 2);
  __bf16* yB    = (__bf16*)take((size_t)Bc * Nn * DIMc * 2);

  hipMemsetAsync(sdeg, 0, (size_t)Bc * H * ANC * sizeof(float), stream);
  k_cvt<<<dim3(2048), 256, 0, stream>>>(Wemb, WembB, TI * DIMc);
  k_cvt<<<dim3(1024), 256, 0, stream>>>(Wout, WoutB, DIMc * DIMc);
  k_cvt<<<dim3(16),   256, 0, stream>>>(Wanc, WancB, ANC * DH);

  k_embed<<<dim3(256, 32), 256, 0, stream>>>(x, WembB, WancB, attnB, vB, sdeg);
  k_aggregate<<<dim3(Bc * H), 128, 0, stream>>>(attnB, vB, sdeg, out2B);
  k_broadcast<<<dim3(Bc * H * (Nn / 128)), 256, 0, stream>>>(attnB, out2B, yB);
  k_output<<<dim3(256, 16), 256, 0, stream>>>(yB, WoutB, bout, out);
}
